// HAN_82454782148686
// MI455X (gfx1250) — matrix operand order repaired and checked by reading, not compiled, not run on hardware
//
#include <hip/hip_runtime.h>
#include <hip/hip_bf16.h>

// ---------------------------------------------------------------------------
// HAN (2-layer HANConv + classifier) for MI455X / gfx1250, wave32 + WMMA.
//
// d_in layout (setup_inputs dict insertion order, params flattened in
// insertion order, (W,b) tuples flattened W-then-b):
//   0 x_transaction  1 x_user  2 x_merchant
//   3 ut_src 4 ut_dst 5 tu_src 6 tu_dst 7 mt_src 8 mt_dst 9 tm_src 10 tm_dst
//   l1: 11 Wt 12 bt 13 Wu 14 bu 15 Wm 16 bm
//       17 aS_ut 18 aD_ut 19 aS_tu 20 aD_tu 21 aS_mt 22 aD_mt 23 aS_tm 24 aD_tm
//       25 kW 26 kb 27 q
//   l2: 28..44 (same structure)
//   cls: 45 Wc[128,2] 46 bc[2]
// ---------------------------------------------------------------------------

#define NT_N 200000
#define NU_N 50000
#define NM_N 20000
#define NE_E 400000
#define HD   128
#define NH   8

typedef __attribute__((ext_vector_type(16))) _Float16 v16h;
typedef __attribute__((ext_vector_type(8)))  float    v8f;

// ---------------------------------------------------------------------------
// WMMA GEMM with fragment-layout LDS:
//   MODE 0: out[N,128] = act(X[N,128] @ W[128,128] + b)   (act: relu flag)
//   MODE 1: colsum[c] += sum_rows tanh((X @ W + b)[r,c])  (matrix never stored)
//
// Block = 128 threads = 4 waves. Each block processes 256 rows (4 chunks of 64).
// W is swizzled once per block into per-(kk,ntile) WMMA B-fragment layout so a
// lane's fragment is 16 contiguous halves -> 2x ds_load_b128. X tiles are
// converted f32->f16 and scattered into A-fragment layout the same way.
// ---------------------------------------------------------------------------
template <int MODE>
__global__ __launch_bounds__(128) void gemm128_kernel(
    const float* __restrict__ X, const float* __restrict__ W,
    const float* __restrict__ Bv, float* __restrict__ outp,
    int Nrows, int actRelu)
{
    __shared__ __align__(32) _Float16 sWf[4 * 8 * 32 * 16]; // [kk][ntile][lane][16]
    __shared__ __align__(32) _Float16 sXf[4 * 32 * 16];     // [wave][lane][16]
    __shared__ float sB[HD];
    __shared__ float sCol[HD];

    const int tid = threadIdx.x;

    // ---- swizzle W (f32 -> f16) into B-fragment layout, one row per thread ----
    {
        int kk   = tid >> 5;          // which K block of 32
        int halfk = (tid >> 4) & 1;   // k within block: 0-15 -> lanes 0-15, 16-31 -> 16-31
        int idx  = tid & 15;          // half-index within lane = k & 15
        const float* wrow = W + (size_t)tid * HD;
        #pragma unroll 4
        for (int ng = 0; ng < HD; ++ng) {
            int ntile = ng >> 4, n = ng & 15;
            int lane  = (halfk << 4) + n;
            sWf[((((kk << 3) + ntile) << 5) + lane) * 16 + idx] = (_Float16)wrow[ng];
        }
    }
    if (tid < HD) { sB[tid] = Bv[tid]; if (MODE == 1) sCol[tid] = 0.f; }

    const int wave = tid >> 5;
    const int lane = tid & 31;
    const int half = lane >> 4;
    const int ln   = lane & 15;

    for (int chunk = 0; chunk < 4; ++chunk) {
        const int row0 = blockIdx.x * 256 + chunk * 64;   // uniform across block
        if (row0 >= Nrows) break;

        v8f acc[8] = {};

        for (int kk = 0; kk < 4; ++kk) {
            __syncthreads();
            {   // load 64x32 X tile, convert f32->f16, scatter into A-fragment layout
                int r  = tid >> 1;
                int c0 = (tid & 1) << 4;
                int gr = row0 + r;
                int wv = r >> 4, lr = r & 15;
                float vals[16];
                if (gr < Nrows) {
                    const float* xp = X + (size_t)gr * HD + (kk << 5) + c0;
                    #pragma unroll
                    for (int c = 0; c < 16; ++c) vals[c] = xp[c];
                    if (gr + 64 < Nrows)   // prefetch next chunk's rows (global_prefetch_b8)
                        __builtin_prefetch(X + (size_t)(gr + 64) * HD + (kk << 5) + c0, 0, 1);
                } else {
                    #pragma unroll
                    for (int c = 0; c < 16; ++c) vals[c] = 0.f;
                }
                #pragma unroll
                for (int c = 0; c < 16; ++c) {
                    int k   = c0 + c;
                    int h2  = k >> 4;
                    int idx = k & 15;
                    sXf[(((wv << 5) + (h2 << 4) + lr) << 4) + idx] = (_Float16)vals[c];
                }
            }
            __syncthreads();

            // fragment loads: contiguous 32B per lane -> ds_load_b128 x2 each
            v16h a = *(const v16h*)&sXf[((wave << 5) + lane) << 4];
            #pragma unroll
            for (int nt = 0; nt < 8; ++nt) {
                v16h b = *(const v16h*)&sWf[((((kk << 3) + nt) << 5) + lane) << 4];
                acc[nt] = __builtin_amdgcn_wmma_f32_16x16x32_f16(
                    false, a, false, b, (short)0, acc[nt], false, false);
            }
        }

        // ---- epilogue for this 64-row chunk ----
        if (MODE == 0) {
            #pragma unroll
            for (int nt = 0; nt < 8; ++nt) {
                #pragma unroll
                for (int r = 0; r < 8; ++r) {
                    int m   = wave * 16 + half * 8 + r;  // D layout: vgpr r -> M=r / M=8+r
                    int gr  = row0 + m;
                    int col = nt * 16 + ln;
                    float v = acc[nt][r] + sB[col];
                    if (actRelu) v = v > 0.f ? v : 0.f;
                    if (gr < Nrows) outp[(size_t)gr * HD + col] = v;
                }
            }
        } else {
            #pragma unroll
            for (int nt = 0; nt < 8; ++nt) {
                #pragma unroll
                for (int r = 0; r < 8; ++r) {
                    int m  = wave * 16 + half * 8 + r;
                    int gr = row0 + m;
                    if (gr < Nrows) {
                        int col = nt * 16 + ln;
                        atomicAdd(&sCol[col], tanhf(acc[nt][r] + sB[col]));
                    }
                }
            }
        }
    }

    if (MODE == 1) {
        __syncthreads();
        if (tid < HD) atomicAdd(&outp[tid], sCol[tid]);
    }
}

// ---------------------------------------------------------------------------
// GAT node / edge kernels (atomic/L2-bound scatter-gather; node tables fit in
// the 192MB L2 so plain global atomics are the right path on MI455X)
// ---------------------------------------------------------------------------
__global__ void node_att_kernel(const float* __restrict__ z,
                                const float* __restrict__ a,
                                float* __restrict__ out, int N)
{
    int t = blockIdx.x * blockDim.x + threadIdx.x;
    if (t >= N * NH) return;
    int n = t >> 3, h = t & 7;
    const float* zp = z + (size_t)n * HD + h * 16;
    const float* ap = a + h * 16;
    float s = 0.f;
    #pragma unroll
    for (int i = 0; i < 16; ++i) s += zp[i] * ap[i];
    out[t] = s;
}

__device__ __forceinline__ unsigned enc_f32(float f) {
    unsigned u = __float_as_uint(f);
    return (u & 0x80000000u) ? ~u : (u | 0x80000000u);
}
__device__ __forceinline__ float dec_f32(unsigned k) {
    return __uint_as_float((k & 0x80000000u) ? (k & 0x7fffffffu) : ~k);
}

__global__ void edge_logits_kernel(const float* __restrict__ as_,
                                   const float* __restrict__ ad_,
                                   const int* __restrict__ src,
                                   const int* __restrict__ dst,
                                   float* __restrict__ ebuf,
                                   unsigned* __restrict__ menc, int E)
{
    int t = blockIdx.x * blockDim.x + threadIdx.x;
    if (t >= E * NH) return;
    int e = t >> 3, h = t & 7;
    float v = as_[src[e] * NH + h] + ad_[dst[e] * NH + h];
    v = v > 0.f ? v : 0.2f * v;               // leaky_relu(., 0.2)
    ebuf[t] = v;
    atomicMax(&menc[dst[e] * NH + h], enc_f32(v));
}

__global__ void edge_exp_kernel(float* __restrict__ ebuf,
                                const unsigned* __restrict__ menc,
                                const int* __restrict__ dst,
                                float* __restrict__ ssum, int E)
{
    int t = blockIdx.x * blockDim.x + threadIdx.x;
    if (t >= E * NH) return;
    int e = t >> 3, h = t & 7;
    float m = dec_f32(menc[dst[e] * NH + h]);
    float p = __expf(ebuf[t] - m);
    ebuf[t] = p;                               // ebuf now holds p
    atomicAdd(&ssum[dst[e] * NH + h], p);
}

__global__ void edge_aggregate_kernel(const float* __restrict__ zsrc,
                                      const int* __restrict__ src,
                                      const int* __restrict__ dst,
                                      const float* __restrict__ pbuf,
                                      const float* __restrict__ ssum,
                                      float* __restrict__ out, int E)
{
    int t = blockIdx.x * blockDim.x + threadIdx.x;   // E*128 = 51.2M fits int
    if (t >= E * HD) return;
    int e = t >> 7, c = t & 127, h = c >> 4;
    int d = dst[e];
    float alpha = pbuf[e * NH + h] / (ssum[d * NH + h] + 1e-16f);
    atomicAdd(&out[(size_t)d * HD + c], zsrc[(size_t)src[e] * HD + c] * alpha);
}

__global__ void relu_inplace_kernel(float* __restrict__ x, int n)
{
    int t = blockIdx.x * blockDim.x + threadIdx.x;
    if (t < n) { float v = x[t]; x[t] = v > 0.f ? v : 0.f; }
}

// ---------------------------------------------------------------------------
// Semantic attention: scores over 2 metapaths -> softmax betas (1 block)
// ---------------------------------------------------------------------------
__global__ __launch_bounds__(128) void semantic_score2_kernel(
    const float* __restrict__ colsum, const float* __restrict__ q,
    float invN, float* __restrict__ beta)
{
    __shared__ float s0[128], s1[128];
    int t = threadIdx.x;
    s0[t] = q[t] * colsum[t] * invN;
    s1[t] = q[t] * colsum[128 + t] * invN;
    __syncthreads();
    for (int st = 64; st > 0; st >>= 1) {
        if (t < st) { s0[t] += s0[t + st]; s1[t] += s1[t + st]; }
        __syncthreads();
    }
    if (t == 0) {
        float a = s0[0], b = s1[0];
        float mx = fmaxf(a, b);
        float ea = __expf(a - mx), eb = __expf(b - mx);
        float inv = 1.f / (ea + eb);
        beta[0] = ea * inv; beta[1] = eb * inv;
    }
}

__global__ void combine2_relu_kernel(const float* __restrict__ O0,
                                     const float* __restrict__ O1,
                                     const float* __restrict__ beta,
                                     float* __restrict__ res, int n)
{
    int t = blockIdx.x * blockDim.x + threadIdx.x;
    if (t >= n) return;
    float v = beta[0] * O0[t] + beta[1] * O1[t];
    res[t] = v > 0.f ? v : 0.f;
}

__global__ void classifier_kernel(const float* __restrict__ Hf,
                                  const float* __restrict__ Wc,
                                  const float* __restrict__ bc,
                                  float* __restrict__ out, int N)
{
    int r = blockIdx.x * blockDim.x + threadIdx.x;
    if (r >= N) return;
    float a0 = bc[0], a1 = bc[1];
    const float* hp = Hf + (size_t)r * HD;
    #pragma unroll 8
    for (int c = 0; c < HD; ++c) {
        float x = hp[c];
        a0 += x * Wc[c * 2 + 0];
        a1 += x * Wc[c * 2 + 1];
    }
    out[r * 2 + 0] = a0;
    out[r * 2 + 1] = a1;
}

// ---------------------------------------------------------------------------
// Host driver
// ---------------------------------------------------------------------------
static inline int cdiv(int a, int b) { return (a + b - 1) / b; }

static void run_gat(const float* zsrc, const float* zdst, int Nsrc, int Ndst,
                    const int* src, const int* dst,
                    const float* aS, const float* aD,
                    float* Obuf, float* asb, float* adb,
                    unsigned* menc, float* ssum, float* ebuf, hipStream_t s)
{
    node_att_kernel<<<cdiv(Nsrc * NH, 256), 256, 0, s>>>(zsrc, aS, asb, Nsrc);
    node_att_kernel<<<cdiv(Ndst * NH, 256), 256, 0, s>>>(zdst, aD, adb, Ndst);
    hipMemsetAsync(menc, 0, (size_t)Ndst * NH * sizeof(unsigned), s);
    hipMemsetAsync(ssum, 0, (size_t)Ndst * NH * sizeof(float), s);
    hipMemsetAsync(Obuf, 0, (size_t)Ndst * HD * sizeof(float), s);
    edge_logits_kernel<<<cdiv(NE_E * NH, 256), 256, 0, s>>>(asb, adb, src, dst, ebuf, menc, NE_E);
    edge_exp_kernel<<<cdiv(NE_E * NH, 256), 256, 0, s>>>(ebuf, menc, dst, ssum, NE_E);
    edge_aggregate_kernel<<<cdiv(NE_E * HD, 256), 256, 0, s>>>(zsrc, src, dst, ebuf, ssum, Obuf, NE_E);
    relu_inplace_kernel<<<cdiv(Ndst * HD, 256), 256, 0, s>>>(Obuf, Ndst * HD);
}

extern "C" void kernel_launch(void* const* d_in, const int* in_sizes, int n_in,
                              void* d_out, int out_size, void* d_ws, size_t ws_size,
                              hipStream_t stream)
{
    (void)in_sizes; (void)n_in; (void)out_size; (void)ws_size;

    const float* xT = (const float*)d_in[0];
    const float* xU = (const float*)d_in[1];
    const float* xM = (const float*)d_in[2];
    const int* ut_s = (const int*)d_in[3]; const int* ut_d = (const int*)d_in[4];
    const int* tu_s = (const int*)d_in[5]; const int* tu_d = (const int*)d_in[6];
    const int* mt_s = (const int*)d_in[7]; const int* mt_d = (const int*)d_in[8];
    const int* tm_s = (const int*)d_in[9]; const int* tm_d = (const int*)d_in[10];

    struct LayerP {
        const float *Wt, *bt, *Wu, *bu, *Wm, *bm;
        const float *aS_ut, *aD_ut, *aS_tu, *aD_tu, *aS_mt, *aD_mt, *aS_tm, *aD_tm;
        const float *kW, *kb, *q;
    };
    auto getL = [&](int base) {
        LayerP L;
        L.Wt = (const float*)d_in[base + 0];  L.bt = (const float*)d_in[base + 1];
        L.Wu = (const float*)d_in[base + 2];  L.bu = (const float*)d_in[base + 3];
        L.Wm = (const float*)d_in[base + 4];  L.bm = (const float*)d_in[base + 5];
        L.aS_ut = (const float*)d_in[base + 6];  L.aD_ut = (const float*)d_in[base + 7];
        L.aS_tu = (const float*)d_in[base + 8];  L.aD_tu = (const float*)d_in[base + 9];
        L.aS_mt = (const float*)d_in[base + 10]; L.aD_mt = (const float*)d_in[base + 11];
        L.aS_tm = (const float*)d_in[base + 12]; L.aD_tm = (const float*)d_in[base + 13];
        L.kW = (const float*)d_in[base + 14]; L.kb = (const float*)d_in[base + 15];
        L.q  = (const float*)d_in[base + 16];
        return L;
    };
    LayerP L1 = getL(11), L2 = getL(28);
    const float* clsW = (const float*)d_in[45];
    const float* clsB = (const float*)d_in[46];

    // ---- workspace partition ----
    char* wsb = (char*)d_ws;
    size_t off = 0;
    auto carve = [&](size_t bytes) -> void* {
        void* p = wsb + off;
        off += (bytes + 255) & ~(size_t)255;
        return p;
    };
    float*    zt   = (float*)carve((size_t)NT_N * HD * 4);
    float*    zu   = (float*)carve((size_t)NU_N * HD * 4);
    float*    zm   = (float*)carve((size_t)NM_N * HD * 4);
    float*    ht   = (float*)carve((size_t)NT_N * HD * 4);
    float*    ot0  = (float*)carve((size_t)NT_N * HD * 4);   // ut -> transaction
    float*    ot1  = (float*)carve((size_t)NT_N * HD * 4);   // mt -> transaction
    float*    ou   = (float*)carve((size_t)NU_N * HD * 4);   // tu -> user
    float*    om   = (float*)carve((size_t)NM_N * HD * 4);   // tm -> merchant
    float*    asb  = (float*)carve((size_t)NT_N * NH * 4);
    float*    adb  = (float*)carve((size_t)NT_N * NH * 4);
    unsigned* menc = (unsigned*)carve((size_t)NT_N * NH * 4);
    float*    ssum = (float*)carve((size_t)NT_N * NH * 4);
    float*    ebuf = (float*)carve((size_t)NE_E * NH * 4);
    float*    csum = (float*)carve(2 * HD * 4);
    float*    beta = (float*)carve(2 * 4);

    // ================= Layer 1 =================
    gemm128_kernel<0><<<cdiv(NT_N, 256), 128, 0, stream>>>(xT, L1.Wt, L1.bt, zt, NT_N, 0);
    gemm128_kernel<0><<<cdiv(NU_N, 256), 128, 0, stream>>>(xU, L1.Wu, L1.bu, zu, NU_N, 0);
    gemm128_kernel<0><<<cdiv(NM_N, 256), 128, 0, stream>>>(xM, L1.Wm, L1.bm, zm, NM_N, 0);

    run_gat(zu, zt, NU_N, NT_N, ut_s, ut_d, L1.aS_ut, L1.aD_ut, ot0, asb, adb, menc, ssum, ebuf, stream);
    run_gat(zm, zt, NM_N, NT_N, mt_s, mt_d, L1.aS_mt, L1.aD_mt, ot1, asb, adb, menc, ssum, ebuf, stream);
    run_gat(zt, zu, NT_N, NU_N, tu_s, tu_d, L1.aS_tu, L1.aD_tu, ou,  asb, adb, menc, ssum, ebuf, stream);
    run_gat(zt, zm, NT_N, NM_N, tm_s, tm_d, L1.aS_tm, L1.aD_tm, om,  asb, adb, menc, ssum, ebuf, stream);

    // semantic attention (transaction, M=2: [ut, mt]); user/merchant M=1 -> beta=1,
    // and O is already ReLU'd, so layer-2 inputs for user/merchant alias ou/om.
    hipMemsetAsync(csum, 0, 2 * HD * sizeof(float), stream);
    gemm128_kernel<1><<<cdiv(NT_N, 256), 128, 0, stream>>>(ot0, L1.kW, L1.kb, csum,      NT_N, 0);
    gemm128_kernel<1><<<cdiv(NT_N, 256), 128, 0, stream>>>(ot1, L1.kW, L1.kb, csum + HD, NT_N, 0);
    semantic_score2_kernel<<<1, 128, 0, stream>>>(csum, L1.q, 1.0f / NT_N, beta);
    combine2_relu_kernel<<<cdiv(NT_N * HD, 256), 256, 0, stream>>>(ot0, ot1, beta, ht, NT_N * HD);

    // ================= Layer 2 (only transaction branch feeds the classifier) ===
    gemm128_kernel<0><<<cdiv(NT_N, 256), 128, 0, stream>>>(ht, L2.Wt, L2.bt, zt, NT_N, 0);
    gemm128_kernel<0><<<cdiv(NU_N, 256), 128, 0, stream>>>(ou, L2.Wu, L2.bu, zu, NU_N, 0);
    gemm128_kernel<0><<<cdiv(NM_N, 256), 128, 0, stream>>>(om, L2.Wm, L2.bm, zm, NM_N, 0);

    run_gat(zu, zt, NU_N, NT_N, ut_s, ut_d, L2.aS_ut, L2.aD_ut, ot0, asb, adb, menc, ssum, ebuf, stream);
    run_gat(zm, zt, NM_N, NT_N, mt_s, mt_d, L2.aS_mt, L2.aD_mt, ot1, asb, adb, menc, ssum, ebuf, stream);

    hipMemsetAsync(csum, 0, 2 * HD * sizeof(float), stream);
    gemm128_kernel<1><<<cdiv(NT_N, 256), 128, 0, stream>>>(ot0, L2.kW, L2.kb, csum,      NT_N, 0);
    gemm128_kernel<1><<<cdiv(NT_N, 256), 128, 0, stream>>>(ot1, L2.kW, L2.kb, csum + HD, NT_N, 0);
    semantic_score2_kernel<<<1, 128, 0, stream>>>(csum, L2.q, 1.0f / NT_N, beta);
    combine2_relu_kernel<<<cdiv(NT_N * HD, 256), 256, 0, stream>>>(ot0, ot1, beta, ht, NT_N * HD);

    // ================= Classifier =================
    classifier_kernel<<<cdiv(NT_N, 256), 256, 0, stream>>>(ht, clsW, clsB, (float*)d_out, NT_N);
}
